// GNN_Agent_35777077576185
// MI455X (gfx1250) — compile-verified, matrix operand order checked by reading, not run on hardware
//
#include <hip/hip_runtime.h>
#include <hip/hip_bf16.h>

#define NN      20000
#define FF      127
#define CC      128
#define LL      4096
#define AAct    32
#define EE      320000
#define BB      8
#define NLAYERS 5
#define ATTD    256
#define BN      (BB * NN)     /* 160000 rows  */
#define MT      (BN / 16)     /* 10000 row tiles (batch boundary aligns: 20000/16=1250) */
#define WST     136           /* padded LDS row stride (halves): 272B = 68 banks -> no conflicts */

#ifndef ASYNC_STAGE
#define ASYNC_STAGE 1         /* stage weights with global_load_async_to_lds_b128 */
#endif

typedef __attribute__((ext_vector_type(16))) _Float16 v16h;
typedef __attribute__((ext_vector_type(8)))  _Float16 v8h;
typedef __attribute__((ext_vector_type(8)))  float    v8f;

// ---------------------------------------------------------------------------
// WMMA operand fragments (wave32, 16-bit A 16x32 / B 32x16 layout):
//   lanes 0-15  hold K = {kb+0..kb+7,  kb+16..kb+23}
//   lanes 16-31 hold K = {kb+8..kb+15, kb+24..kb+31}
// frag_at() works on any address space (global b128 or LDS ds_load_b128 after
// inlining); rows must be 16B aligned.
// ---------------------------------------------------------------------------
__device__ __forceinline__ v16h frag_at(const _Float16* p) {
    v8h lo = *(const v8h*)(p);
    v8h hi = *(const v8h*)(p + 16);
    v16h f;
#pragma unroll
    for (int i = 0; i < 8; ++i) { f[i] = lo[i]; f[i + 8] = hi[i]; }
    return f;
}

__device__ __forceinline__ v16h load_frag(const _Float16* __restrict__ base,
                                          int stride, int row, int kb, int lane) {
    return frag_at(base + (long)row * stride + kb + ((lane >> 4) << 3));
}

__device__ __forceinline__ v8f wmma_f16(v16h a, v16h b, v8f c) {
    return __builtin_amdgcn_wmma_f32_16x16x32_f16(false, a, false, b,
                                                  (short)0, c, false, false);
}

__device__ __forceinline__ float sigmoidf_(float x) { return 1.0f / (1.0f + __expf(-x)); }

// Issue one 16B async global->LDS transfer (ASYNCcnt-tracked path).
__device__ __forceinline__ void async_to_lds_b128(const _Float16* gsrc, _Float16* lds_dst) {
#if ASYNC_STAGE
    unsigned laddr = (unsigned)(size_t)lds_dst;   // low 32 bits of generic = LDS offset
    asm volatile("global_load_async_to_lds_b128 %0, %1, off"
                 :: "v"(laddr), "v"(gsrc) : "memory");
#else
    *(v8h*)lds_dst = *(const v8h*)gsrc;
#endif
}

__device__ __forceinline__ void wait_async_stage() {
#if ASYNC_STAGE
#if __has_builtin(__builtin_amdgcn_s_wait_asynccnt)
    __builtin_amdgcn_s_wait_asynccnt(0);
#else
    asm volatile("s_wait_asynccnt 0x0" ::: "memory");
#endif
#endif
}

// ---------------------------------------------------------------------------
// Setup kernels
// ---------------------------------------------------------------------------
__global__ __launch_bounds__(256) void cvt_conv_w(const float* __restrict__ w,
                                                  _Float16* __restrict__ out) {
    int idx = blockIdx.x * 256 + threadIdx.x;           // 5*128*128
    if (idx >= NLAYERS * CC * CC) return;
    int i = idx >> 14, r = idx & 16383;
    int n = r >> 7, k = r & 127;
    out[idx] = (_Float16)w[(i << 14) + (k << 7) + n];   // out[i][n][k] = w[i][k][n]
}

__global__ __launch_bounds__(256) void cvt_direct(const float* __restrict__ w,
                                                  _Float16* __restrict__ out, int n) {
    int idx = blockIdx.x * 256 + threadIdx.x;
    if (idx < n) out[idx] = (_Float16)w[idx];
}

__global__ __launch_bounds__(256) void build_ni(const float* __restrict__ nodes,
                                                float* __restrict__ ni32,
                                                _Float16* __restrict__ ni16) {
    long idx = (long)blockIdx.x * 256 + threadIdx.x;    // BN*128
    if (idx >= (long)BN * CC) return;
    int c = (int)(idx & 127);
    long row = idx >> 7;
    int n = (int)(row % NN);
    float v = (c < FF) ? nodes[(long)n * FF + c] : 0.0f;
    ni32[idx] = v;
    ni16[idx] = (_Float16)v;
}

__global__ __launch_bounds__(256) void scatter_cov(const float* __restrict__ input,
                                                   const int* __restrict__ c2l,
                                                   float* __restrict__ ni32,
                                                   _Float16* __restrict__ ni16) {
    int idx = blockIdx.x * 256 + threadIdx.x;           // B*L
    if (idx >= BB * LL) return;
    int b = idx / LL, l = idx % LL;
    int node = c2l[l];
    float v = input[(long)b * (LL + AAct) + l];
    long off = ((long)b * NN + node) * CC + (CC - 1);
    ni32[off] = v;
    ni16[off] = (_Float16)v;
}

__global__ __launch_bounds__(256) void zero_f32(float* __restrict__ p, long n) {
    long idx = (long)blockIdx.x * 256 + threadIdx.x;
    if (idx < n) p[idx] = 0.0f;
}

__global__ __launch_bounds__(256) void cvt_f32_f16(const float* __restrict__ in,
                                                   _Float16* __restrict__ out, long n) {
    long idx = (long)blockIdx.x * 256 + threadIdx.x;
    if (idx < n) out[idx] = (_Float16)in[idx];
}

// ---------------------------------------------------------------------------
// Conv GEMM: m[BN x 128] = x16 @ Wt(layout [n][k]). Weights staged once per
// block into LDS (async), B fragments served by ds_load_b128; one wave = full
// 16x128 row strip (8 column tiles), A fragment loaded once per k-step.
// ---------------------------------------------------------------------------
__global__ __launch_bounds__(256) void conv_gemm(const _Float16* __restrict__ A,
                                                 const _Float16* __restrict__ Wt,
                                                 float* __restrict__ OUT) {
    __shared__ __align__(16) _Float16 wlds[CC * WST];
    int t = threadIdx.x;
    for (int i = t; i < CC * 16; i += 256) {            // 2048 x 16B chunks
        int row = i >> 4, ch = i & 15;
        async_to_lds_b128(Wt + row * CC + ch * 8, &wlds[row * WST + ch * 8]);
    }
    wait_async_stage();
    __syncthreads();

    int wave = blockIdx.x * 8 + (t >> 5);               // grid exact: MT waves
    int lane = t & 31;
    int arow = wave * 16 + (lane & 15);
    int lc = lane & 15;
    int khi = (lane >> 4) << 3;
    v8f zero = {};
    v8f acc[8];
#pragma unroll
    for (int tn = 0; tn < 8; ++tn) acc[tn] = zero;
#pragma unroll
    for (int kb = 0; kb < CC; kb += 32) {
        v16h a = load_frag(A, CC, arow, kb, lane);
#pragma unroll
        for (int tn = 0; tn < 8; ++tn) {
            v16h b = frag_at(&wlds[(tn * 16 + lc) * WST + kb + khi]);
            acc[tn] = wmma_f16(a, b, acc[tn]);
        }
    }
    int rbase = wave * 16 + khi;
#pragma unroll
    for (int tn = 0; tn < 8; ++tn)
#pragma unroll
        for (int r = 0; r < 8; ++r)
            OUT[(long)(rbase + r) * CC + tn * 16 + lc] = acc[tn][r];
}

// ---------------------------------------------------------------------------
// Edge scatter: agg[b*N+dst] += m[b*N+src]; one wave per (batch, edge)
// ---------------------------------------------------------------------------
__global__ __launch_bounds__(256) void edge_scatter(const int* __restrict__ edges,
                                                    const float* __restrict__ m,
                                                    float* __restrict__ agg) {
    long wave = (long)blockIdx.x * 8 + (threadIdx.x >> 5);
    int lane = threadIdx.x & 31;
    if (wave >= (long)BB * EE) return;
    int b = (int)(wave / EE);
    int e = (int)(wave % EE);
    int s = edges[e];
    int d = edges[EE + e];
    long srow = ((long)b * NN + s) * CC;
    long drow = ((long)b * NN + d) * CC;
    float4 v = *(const float4*)(m + srow + lane * 4);
    float* dp = agg + drow + lane * 4;
    atomicAdd(dp + 0, v.x);
    atomicAdd(dp + 1, v.y);
    atomicAdd(dp + 2, v.z);
    atomicAdd(dp + 3, v.w);
}

// ---------------------------------------------------------------------------
// Fully fused GRU layer update (gate GEMMs + nonlinearity + state blend):
//   r = sig(agg@Wi_r^T + x@Wh_r^T + br)   -> one accumulator, 2 chained WMMAs
//   z = sig(... z rows ...)               -> one accumulator, 2 chained WMMAs
//   n = tanh(i_n + r * h_n)               -> i_n, h_n kept in separate accs
//   x' = (1-z)*n + z*x                    -> read xc32, write xn32/xn16
// Block = (row-tile group) x (2 output column tiles tc = 2*tcg, 2*tcg+1).
// LDS stages the 6x32 needed Wi/Wh rows (192 rows, 52KB) via async loads.
// Eliminates the 654 MB/layer gates round-trip through HBM.
// ---------------------------------------------------------------------------
__global__ __launch_bounds__(256) void gru_fused(const _Float16* __restrict__ agg16,
                                                 const _Float16* __restrict__ xc16,
                                                 const float* __restrict__ xc32,
                                                 const _Float16* __restrict__ wi,
                                                 const _Float16* __restrict__ wh,
                                                 const float* __restrict__ bi,
                                                 const float* __restrict__ bh,
                                                 float* __restrict__ xn32,
                                                 _Float16* __restrict__ xn16) {
    // LDS rows: [m(wi=0,wh=1)][gate seg s(r=0,z=1,n=2)][32 rows of col block]
    __shared__ __align__(16) _Float16 wlds[192 * WST];
    int tcg = blockIdx.x & 3;                           // column block: cols 32*tcg..+31
    int tmb = blockIdx.x >> 2;
    int t = threadIdx.x;
    for (int i = t; i < 192 * 16; i += 256) {           // 3072 x 16B chunks
        int row = i >> 4, ch = i & 15;
        int m = row / 96, rs = row % 96;
        int s = rs >> 5, rloc = rs & 31;
        const _Float16* src = (m ? wh : wi) + (s * 128 + tcg * 32 + rloc) * CC + ch * 8;
        async_to_lds_b128(src, &wlds[row * WST + ch * 8]);
    }
    wait_async_stage();
    __syncthreads();

    int w = t >> 5, lane = t & 31;
    int tm = tmb * 8 + w;
    int arow = tm * 16 + (lane & 15);
    int lc = lane & 15;
    int khi = (lane >> 4) << 3;
    v8f zero = {};
    v8f ar[2], az[2], ain[2], ahn[2];
#pragma unroll
    for (int j = 0; j < 2; ++j) { ar[j] = zero; az[j] = zero; ain[j] = zero; ahn[j] = zero; }
#pragma unroll
    for (int kb = 0; kb < CC; kb += 32) {
        v16h a1 = load_frag(agg16, CC, arow, kb, lane);   // input-gate operand
        v16h a2 = load_frag(xc16,  CC, arow, kb, lane);   // hidden-gate operand
#pragma unroll
        for (int j = 0; j < 2; ++j) {
            int rloc = j * 16 + lc;
            int o = kb + khi;
            ar[j]  = wmma_f16(a1, frag_at(&wlds[(rloc)            * WST + o]), ar[j]);
            ar[j]  = wmma_f16(a2, frag_at(&wlds[(96 + rloc)       * WST + o]), ar[j]);
            az[j]  = wmma_f16(a1, frag_at(&wlds[(32 + rloc)       * WST + o]), az[j]);
            az[j]  = wmma_f16(a2, frag_at(&wlds[(96 + 32 + rloc)  * WST + o]), az[j]);
            ain[j] = wmma_f16(a1, frag_at(&wlds[(64 + rloc)       * WST + o]), ain[j]);
            ahn[j] = wmma_f16(a2, frag_at(&wlds[(96 + 64 + rloc)  * WST + o]), ahn[j]);
        }
    }
    int rbase = tm * 16 + khi;
#pragma unroll
    for (int j = 0; j < 2; ++j) {
        int col = (tcg * 2 + j) * 16 + lc;
        float br  = bi[col] + bh[col];
        float bz  = bi[CC + col] + bh[CC + col];
        float bin = bi[2 * CC + col];
        float bhn = bh[2 * CC + col];
#pragma unroll
        for (int r = 0; r < 8; ++r) {
            float rr = sigmoidf_(ar[j][r] + br);
            float zz = sigmoidf_(az[j][r] + bz);
            float nv = tanhf(ain[j][r] + bin + rr * (ahn[j][r] + bhn));
            long o = (long)(rbase + r) * CC + col;
            float h = xc32[o];
            float out = (1.0f - zz) * nv + zz * h;
            xn32[o] = out;
            xn16[o] = (_Float16)out;
        }
    }
}

// ---------------------------------------------------------------------------
// MLP GEMM with relu, f16 output into cat buffer cols [0,128); LDS-staged
// weights; wave = 16x128 strip
// ---------------------------------------------------------------------------
__global__ __launch_bounds__(256) void mlp_gemm_cat(const _Float16* __restrict__ A,
                                                    const _Float16* __restrict__ Wt,
                                                    const float* __restrict__ bias,
                                                    _Float16* __restrict__ cat) {
    __shared__ __align__(16) _Float16 wlds[CC * WST];
    int t = threadIdx.x;
    for (int i = t; i < CC * 16; i += 256) {
        int row = i >> 4, ch = i & 15;
        async_to_lds_b128(Wt + row * CC + ch * 8, &wlds[row * WST + ch * 8]);
    }
    wait_async_stage();
    __syncthreads();

    int wave = blockIdx.x * 8 + (t >> 5);
    int lane = t & 31;
    int arow = wave * 16 + (lane & 15);
    int lc = lane & 15;
    int khi = (lane >> 4) << 3;
    v8f zero = {};
    v8f acc[8];
#pragma unroll
    for (int tn = 0; tn < 8; ++tn) acc[tn] = zero;
#pragma unroll
    for (int kb = 0; kb < CC; kb += 32) {
        v16h a = load_frag(A, CC, arow, kb, lane);
#pragma unroll
        for (int tn = 0; tn < 8; ++tn) {
            v16h b = frag_at(&wlds[(tn * 16 + lc) * WST + kb + khi]);
            acc[tn] = wmma_f16(a, b, acc[tn]);
        }
    }
    int rbase = wave * 16 + khi;
#pragma unroll
    for (int tn = 0; tn < 8; ++tn) {
        float bv = bias[tn * 16 + lc];
#pragma unroll
        for (int r = 0; r < 8; ++r) {
            float v = acc[tn][r] + bv;
            cat[(long)(rbase + r) * (2 * CC) + tn * 16 + lc] =
                (_Float16)(v > 0.0f ? v : 0.0f);
        }
    }
}

__global__ __launch_bounds__(256) void fill_cat(const _Float16* __restrict__ ni16,
                                                _Float16* __restrict__ cat) {
    long idx = (long)blockIdx.x * 256 + threadIdx.x;
    if (idx >= (long)BN * CC) return;
    long m = idx >> 7;
    int c = (int)(idx & 127);
    cat[m * (2 * CC) + CC + c] = ni16[idx];
}

// ---------------------------------------------------------------------------
// Attention + reduction: gbuf[b][col] += sum_rows sigmoid(cat@ai^T+bi)*relu(cat@aj^T+bj)
// grid = B * 16 coltiles * 16 groups; 8 waves/block; each wave strides row tiles.
// All 16 weight fragments (8 k-steps x {ai,aj}) are hoisted into registers, so the
// row loop issues only A loads + 16 WMMAs per tile. Next tile is prefetched.
// ---------------------------------------------------------------------------
__global__ __launch_bounds__(256) void attn_reduce(const _Float16* __restrict__ cat,
                                                   const _Float16* __restrict__ ai,
                                                   const _Float16* __restrict__ aj,
                                                   const float* __restrict__ aib,
                                                   const float* __restrict__ ajb,
                                                   float* __restrict__ gbuf) {
    int bid = blockIdx.x;
    int b    = bid >> 8;
    int tn   = (bid >> 4) & 15;
    int gidx = bid & 15;
    int w = threadIdx.x >> 5, lane = threadIdx.x & 31;
    int wv = gidx * 8 + w;                          // 0..127
    int col = tn * 16 + (lane & 15);
    float bi = aib[col], bj = ajb[col];

    v16h bif[8], bjf[8];                            // hoisted weight fragments
#pragma unroll
    for (int k = 0; k < 8; ++k) {
        bif[k] = load_frag(ai, 2 * CC, col, k * 32, lane);
        bjf[k] = load_frag(aj, 2 * CC, col, k * 32, lane);
    }

    float acc = 0.0f;
    for (int rt = wv; rt < NN / 16; rt += 128) {    // 1250 row tiles / batch
        int arow = b * NN + rt * 16 + (lane & 15);
        if (rt + 128 < NN / 16)                     // prefetch next strided tile
            __builtin_prefetch(cat + ((long)(arow + 128 * 16)) * (2 * CC), 0, 1);
        v8f ci = {}, cj = {};
#pragma unroll
        for (int k = 0; k < 8; ++k) {
            v16h a = load_frag(cat, 2 * CC, arow, k * 32, lane);
            ci = wmma_f16(a, bif[k], ci);
            cj = wmma_f16(a, bjf[k], cj);
        }
#pragma unroll
        for (int r = 0; r < 8; ++r) {
            float si = sigmoidf_(ci[r] + bi);
            float rj = cj[r] + bj;
            acc += si * (rj > 0.0f ? rj : 0.0f);
        }
    }
    atomicAdd(&gbuf[b * ATTD + col], acc);
}

// ---------------------------------------------------------------------------
// Final head: g=relu(gbuf); value = g@vw+vb; adv = g@aw^T+ab; out = value+adv-mean(adv)
// ---------------------------------------------------------------------------
__global__ __launch_bounds__(256) void final_head(const float* __restrict__ gbuf,
                                                  const float* __restrict__ vw,
                                                  const float* __restrict__ vb,
                                                  const float* __restrict__ aw,
                                                  const float* __restrict__ ab,
                                                  float* __restrict__ out) {
    __shared__ float sg[BB * ATTD];
    __shared__ float sred[256];
    int t = threadIdx.x;
    for (int i = t; i < BB * ATTD; i += 256) {
        float v = gbuf[i];
        sg[i] = v > 0.0f ? v : 0.0f;
    }
    __syncthreads();
    int b = t >> 5, a = t & 31;
    float adv = ab[a];
    float val = vb[0];
    for (int k = 0; k < ATTD; ++k) {
        float gv = sg[b * ATTD + k];
        adv += gv * aw[a * ATTD + k];
        val += gv * vw[k];
    }
    sred[t] = adv;
    __syncthreads();
    for (int s = 128; s > 0; s >>= 1) {
        if (t < s) sred[t] += sred[t + s];
        __syncthreads();
    }
    float mean = sred[0] / 256.0f;
    out[t] = val + adv - mean;
}

// ---------------------------------------------------------------------------
// Host launcher
// ---------------------------------------------------------------------------
extern "C" void kernel_launch(void* const* d_in, const int* in_sizes, int n_in,
                              void* d_out, int out_size, void* d_ws, size_t ws_size,
                              hipStream_t stream) {
    const float* input   = (const float*)d_in[0];
    const float* nodes   = (const float*)d_in[1];
    const int*   edges   = (const int*)  d_in[2];
    const int*   c2l     = (const int*)  d_in[3];
    const float* conv_w  = (const float*)d_in[4];
    const float* gru_wi  = (const float*)d_in[5];
    const float* gru_wh  = (const float*)d_in[6];
    const float* gru_bi  = (const float*)d_in[7];
    const float* gru_bh  = (const float*)d_in[8];
    const float* mlp_w   = (const float*)d_in[9];
    const float* mlp_b   = (const float*)d_in[10];
    const float* ai_w    = (const float*)d_in[11];
    const float* ai_b    = (const float*)d_in[12];
    const float* aj_w    = (const float*)d_in[13];
    const float* aj_b    = (const float*)d_in[14];
    const float* v_w     = (const float*)d_in[15];
    const float* v_b     = (const float*)d_in[16];
    const float* adv_w   = (const float*)d_in[17];
    const float* adv_b   = (const float*)d_in[18];
    float* out = (float*)d_out;

    // --- workspace layout ---
    char* ws = (char*)d_ws;
    size_t off = 0;
    auto take = [&](size_t bytes) {
        size_t r = off;
        off = (off + bytes + 255) & ~(size_t)255;
        return r;
    };
    const size_t nBN128f = (size_t)BN * CC * sizeof(float);      // 81.92 MB
    const size_t nBN128h = (size_t)BN * CC * sizeof(_Float16);   // 40.96 MB
    float*    ni32  = (float*)   (ws + take(nBN128f));
    _Float16* ni16  = (_Float16*)(ws + take(nBN128h));
    float*    xa32  = (float*)   (ws + take(nBN128f));
    _Float16* xa16  = (_Float16*)(ws + take(nBN128h));
    float*    xb32  = (float*)   (ws + take(nBN128f));
    _Float16* xb16  = (_Float16*)(ws + take(nBN128h));
    float*    m32   = (float*)   (ws + take(nBN128f));
    float*    agg32 = (float*)   (ws + take(nBN128f));
    _Float16* agg16 = (_Float16*)(ws + take(nBN128h));
    _Float16* cat16 = (_Float16*)(ws + take((size_t)BN * 2 * CC * sizeof(_Float16)));
    float*    gbuf  = (float*)   (ws + take((size_t)BB * ATTD * sizeof(float)));
    _Float16* cw16  = (_Float16*)(ws + take((size_t)NLAYERS * CC * CC * sizeof(_Float16)));
    _Float16* wi16  = (_Float16*)(ws + take((size_t)3 * CC * CC * sizeof(_Float16)));
    _Float16* wh16  = (_Float16*)(ws + take((size_t)3 * CC * CC * sizeof(_Float16)));
    _Float16* mlp16 = (_Float16*)(ws + take((size_t)CC * CC * sizeof(_Float16)));
    _Float16* ai16  = (_Float16*)(ws + take((size_t)ATTD * 2 * CC * sizeof(_Float16)));
    _Float16* aj16  = (_Float16*)(ws + take((size_t)ATTD * 2 * CC * sizeof(_Float16)));
    (void)ws_size; (void)in_sizes; (void)n_in; (void)out_size;

    const int ELEM_BLK = (BN * CC) / 256;           // 80000

    // --- weight conversion (f32 -> f16, output-major [n][k]) ---
    cvt_conv_w<<<(NLAYERS * CC * CC + 255) / 256, 256, 0, stream>>>(conv_w, cw16);
    cvt_direct<<<(3 * CC * CC + 255) / 256, 256, 0, stream>>>(gru_wi, wi16, 3 * CC * CC);
    cvt_direct<<<(3 * CC * CC + 255) / 256, 256, 0, stream>>>(gru_wh, wh16, 3 * CC * CC);
    cvt_direct<<<(CC * CC + 255) / 256, 256, 0, stream>>>(mlp_w, mlp16, CC * CC);
    cvt_direct<<<(ATTD * 2 * CC + 255) / 256, 256, 0, stream>>>(ai_w, ai16, ATTD * 2 * CC);
    cvt_direct<<<(ATTD * 2 * CC + 255) / 256, 256, 0, stream>>>(aj_w, aj16, ATTD * 2 * CC);

    // --- input assembly (layer 0 state is ni directly; no copy needed) ---
    build_ni<<<ELEM_BLK, 256, 0, stream>>>(nodes, ni32, ni16);
    scatter_cov<<<(BB * LL) / 256, 256, 0, stream>>>(input, c2l, ni32, ni16);

    // --- GNN layers (ping-pong state buffers) ---
    const float*    cur32 = ni32;
    const _Float16* cur16 = ni16;
    for (int layer = 0; layer < NLAYERS; ++layer) {
        conv_gemm<<<MT / 8, 256, 0, stream>>>(cur16, cw16 + (size_t)layer * CC * CC, m32);
        zero_f32<<<ELEM_BLK, 256, 0, stream>>>(agg32, (long)BN * CC);
        edge_scatter<<<(BB * EE) / 8, 256, 0, stream>>>(edges, m32, agg32);
        cvt_f32_f16<<<ELEM_BLK, 256, 0, stream>>>(agg32, agg16, (long)BN * CC);
        float*    n32 = (layer & 1) ? xb32 : xa32;
        _Float16* n16 = (layer & 1) ? xb16 : xa16;
        gru_fused<<<(MT / 8) * 4, 256, 0, stream>>>(agg16, cur16, cur32,
                                                    wi16, wh16, gru_bi, gru_bh,
                                                    n32, n16);
        cur32 = n32;
        cur16 = n16;
    }

    // --- readout ---
    mlp_gemm_cat<<<MT / 8, 256, 0, stream>>>(cur16, mlp16, mlp_b, cat16);
    fill_cat<<<ELEM_BLK, 256, 0, stream>>>(ni16, cat16);
    zero_f32<<<(BB * ATTD + 255) / 256, 256, 0, stream>>>(gbuf, BB * ATTD);
    attn_reduce<<<BB * 16 * 16, 256, 0, stream>>>(cat16, ai16, aj16, ai_b, aj_b, gbuf);
    final_head<<<1, 256, 0, stream>>>(gbuf, v_w, v_b, adv_w, adv_b, out);
}
